// MultiHeadAttention_46763603919343
// MI455X (gfx1250) — matrix operand order repaired and checked by reading, not compiled, not run on hardware
//
#include <hip/hip_runtime.h>
#include <hip/hip_bf16.h>
#include <math.h>

typedef __bf16 bfx8  __attribute__((ext_vector_type(8)));
typedef __bf16 bfx16 __attribute__((ext_vector_type(16)));
typedef float  fx8   __attribute__((ext_vector_type(8)));

#define BATCH  4
#define SEQ    2048
#define DMODEL 1024
#define NHEAD  16
#define HDIM   64
#define MTOT   (BATCH * SEQ)   // 8192 tokens

// ---------------------------------------------------------------------------
// WMMA fragment helpers (CDNA5 wave32, v_wmma_f32_16x16x32_bf16)
// A (16x32, bf16): lane m = lane&15; half h = lane>>4;
//   elems 0..15 = K (k0+16h) .. (k0+16h+15)
// B (32x16, bf16): lane n = lane&15; elems 0..15 = K (k0+16h) .. (k0+16h+15)
// C/D (16x16, f32): c[r] -> row (r + 8h), col (lane&15)
// ---------------------------------------------------------------------------
static __device__ __forceinline__ bfx16 load_fragA(const __bf16* rowbase, int k0, int half) {
  union { bfx16 v; bfx8 h[2]; } u;
  u.h[0] = *(const bfx8*)(rowbase + k0 + 16 * half);
  u.h[1] = *(const bfx8*)(rowbase + k0 + 16 * half + 8);
  return u.v;
}
static __device__ __forceinline__ bfx16 load_fragB(const __bf16* colbase, int k0, int half) {
  const __bf16* p = colbase + k0 + 16 * half;
  union { bfx16 v; bfx8 h[2]; } u;
  u.h[0] = *(const bfx8*)(p);
  u.h[1] = *(const bfx8*)(p + 8);
  return u.v;
}
static __device__ __forceinline__ fx8 wmma_bf16(bfx16 a, bfx16 b, fx8 c) {
  return __builtin_amdgcn_wmma_f32_16x16x32_bf16(false, a, false, b, (short)0, c, false, false);
}

// Generic pointer -> 32-bit LDS byte offset (for async-to-LDS vdst operand).
typedef __attribute__((address_space(3))) char lds_char;
static __device__ __forceinline__ unsigned lds_addr_of(const void* p) {
  return (unsigned)(unsigned long long)(lds_char*)const_cast<void*>(p);
}

// ---------------------------------------------------------------------------
// Kernel 1a: fp32 -> bf16 (X)
// ---------------------------------------------------------------------------
__global__ __launch_bounds__(256) void cvt_f32_bf16(const float* __restrict__ in,
                                                    __bf16* __restrict__ out, int n) {
  int i = (blockIdx.x * 256 + threadIdx.x) * 4;
  if (i < n) {
    float4 v = *(const float4*)(in + i);
    union { struct { __bf16 e[4]; } s; unsigned long long u; } pk;
    pk.s.e[0] = (__bf16)v.x; pk.s.e[1] = (__bf16)v.y;
    pk.s.e[2] = (__bf16)v.z; pk.s.e[3] = (__bf16)v.w;
    *(unsigned long long*)(out + i) = pk.u;
  }
}

// Kernel 1b: W[K][N] fp32 -> Wt[N][K] bf16  (so B-fragments are K-contiguous)
__global__ __launch_bounds__(256) void transpose_w(const float* __restrict__ W,
                                                   __bf16* __restrict__ Wt, int K, int N) {
  int idx = blockIdx.x * 256 + threadIdx.x;   // output-major over N*K
  if (idx < N * K) {
    int n = idx / K, k = idx - n * K;
    Wt[idx] = (__bf16)W[(size_t)k * N + n];
  }
}

// ---------------------------------------------------------------------------
// Kernel 2/4: bf16 WMMA GEMM  C[M][N] = A[M][K] * Bt[N][K]^T   (f32 accum)
// Wave tile 32x64 (2x4 of 16x16), 8 waves/block -> block tile 128x128.
// B tile (128 cols x 32 K) staged into LDS with GLOBAL_LOAD_ASYNC_TO_LDS_B128,
// double-buffered: DMA for K-tile t+1 overlaps the 8 WMMAs of tile t.
// MODE 0: store bf16, [B,H,S,Dh] layout (Q,K)      (gm->(b,s), gn->(h,d))
// MODE 1: store bf16, [B,H,Dh,S] layout (V^T)
// MODE 2: store f32 row-major + bias (output projection)
// ---------------------------------------------------------------------------
template <int MODE>
__global__ __launch_bounds__(256) void gemm_bf16(const __bf16* __restrict__ A,
                                                 const __bf16* __restrict__ Bt,
                                                 void* __restrict__ out,
                                                 const float* __restrict__ bias,
                                                 int M, int N, int K) {
  // Row stride 40 elems (80 B): 64 B of K-data + pad, 16-B aligned rows.
  __shared__ __align__(16) __bf16 sB[2][128][40];   // 20 KB

  const int tid  = threadIdx.x;
  const int wave = tid >> 5;
  const int lane = tid & 31;
  const int half = lane >> 4;
  const int ln   = lane & 15;
  const int wm = wave & 3, wn = wave >> 2;
  const int m0   = blockIdx.y * 128 + wm * 32;
  const int nblk = blockIdx.x * 128;

  const __bf16* arow[2];
#pragma unroll
  for (int i = 0; i < 2; ++i) arow[i] = A + (size_t)(m0 + 16 * i + ln) * K;

  // Staging assignment: 128 rows x 4 16-byte chunks = 512 chunks, 2 per thread.
  const int srow = tid >> 1;            // 0..127
  const int q2   = (tid & 1) * 2;       // chunks q2, q2+1
  const __bf16* gB = Bt + (size_t)(nblk + srow) * K + q2 * 8;

  auto stage = [&](int buf, int k0) {
    const __bf16* g = gB + k0;
    unsigned l = lds_addr_of(&sB[buf][srow][q2 * 8]);
    asm volatile("global_load_async_to_lds_b128 %0, %1, off"
                 :: "v"(l), "v"(g) : "memory");
    asm volatile("global_load_async_to_lds_b128 %0, %1, off"
                 :: "v"(l + 16u), "v"(g + 8) : "memory");
  };

  stage(0, 0);
  fx8 acc[2][4] = {};
  const int KT = K >> 5;
  for (int t = 0; t < KT; ++t) {
    const int k0  = t << 5;
    const int buf = t & 1;

    bfx16 af[2];
#pragma unroll
    for (int i = 0; i < 2; ++i) af[i] = load_fragA(arow[i], k0, half);

    asm volatile("s_wait_asynccnt 0x0" ::: "memory");  // this K-tile's DMA done
    __syncthreads();                                   // visible to all waves
    if (t + 1 < KT) stage(buf ^ 1, k0 + 32);           // prefetch next K-tile

    bfx16 bf[4];
#pragma unroll
    for (int j = 0; j < 4; ++j)
      bf[j] = load_fragB(&sB[buf][wn * 64 + 16 * j + ln][0], 0, half);
#pragma unroll
    for (int i = 0; i < 2; ++i)
#pragma unroll
      for (int j = 0; j < 4; ++j) acc[i][j] = wmma_bf16(af[i], bf[j], acc[i][j]);
  }

  const int n0 = nblk + wn * 64;
#pragma unroll
  for (int i = 0; i < 2; ++i) {
#pragma unroll
    for (int j = 0; j < 4; ++j) {
#pragma unroll
      for (int r = 0; r < 8; ++r) {
        int gm = m0 + 16 * i + 8 * half + r;
        int gn = n0 + 16 * j + ln;
        float v = acc[i][j][r];
        if constexpr (MODE == 2) {
          ((float*)out)[(size_t)gm * N + gn] = v + bias[gn];
        } else {
          int b = gm >> 11, s = gm & (SEQ - 1);   // SEQ = 2048
          int h = gn >> 6,  d = gn & (HDIM - 1);  // HDIM = 64
          if constexpr (MODE == 0) {
            ((__bf16*)out)[((((size_t)b * NHEAD + h) * SEQ + s) << 6) + d] = (__bf16)v;
          } else {
            ((__bf16*)out)[(((size_t)b * NHEAD + h) * HDIM + d) * SEQ + s] = (__bf16)v;
          }
        }
      }
    }
  }
}

// ---------------------------------------------------------------------------
// Kernel 3: causal flash attention. One wave per (b,h,16-query tile).
// Key tiles of 64; scores via WMMA -> LDS -> online softmax (f32) -> P (bf16)
// -> WMMA P*V with f32 accum; output bf16 to C [B,S,H*Dh].
// ---------------------------------------------------------------------------
__global__ __launch_bounds__(128) void attn_causal(const __bf16* __restrict__ Qb,
                                                   const __bf16* __restrict__ Kb,
                                                   const __bf16* __restrict__ Vt,
                                                   __bf16* __restrict__ C) {
  __shared__ float                sS[4][16][68];
  __shared__ __align__(16) __bf16 sP[4][16][72];
  __shared__ float                sScale[4][16];

  const int wv   = threadIdx.x >> 5;
  const int lane = threadIdx.x & 31;
  const int half = lane >> 4;
  const int ln   = lane & 15;

  const int wid = blockIdx.x * 4 + wv;   // 0 .. 8191
  const int bh  = wid >> 7;              // (b*16+h), 0..63
  const int qt  = wid & 127;
  const int b   = bh >> 4;
  const int h   = bh & 15;
  const int q0  = qt << 4;

  const __bf16* Qh  = Qb + (size_t)bh * SEQ * HDIM;
  const __bf16* Kh  = Kb + (size_t)bh * SEQ * HDIM;
  const __bf16* Vth = Vt + (size_t)bh * HDIM * SEQ;

  // Q fragments for this query tile are loop-invariant.
  const __bf16* qrow = Qh + (size_t)(q0 + ln) * HDIM;
  const bfx16 aq0 = load_fragA(qrow, 0, half);
  const bfx16 aq1 = load_fragA(qrow, 32, half);

  fx8 o[4] = {};
  float m_i = -INFINITY;   // valid in lanes 0..15 (row = lane)
  float l_i = 0.0f;

  const int nkt = (q0 + 16 + 63) >> 6;   // causal: keys 0 .. q0+15
  for (int kt = 0; kt < nkt; ++kt) {
    const int kk0 = kt << 6;

    // ---- scores S = (Q K^T) / 8, masked, into LDS ----
#pragma unroll
    for (int nt = 0; nt < 4; ++nt) {
      const __bf16* kcol = Kh + (size_t)(kk0 + 16 * nt + ln) * HDIM;
      bfx16 bk0 = load_fragB(kcol, 0, half);
      bfx16 bk1 = load_fragB(kcol, 32, half);
      fx8 c = {};
      c = wmma_bf16(aq0, bk0, c);
      c = wmma_bf16(aq1, bk1, c);
      const int key = kk0 + 16 * nt + ln;
#pragma unroll
      for (int r = 0; r < 8; ++r) {
        int q = q0 + 8 * half + r;
        float sv = c[r] * 0.125f;                       // 1/sqrt(64)
        sS[wv][8 * half + r][16 * nt + ln] = (key <= q) ? sv : -INFINITY;
      }
    }
    asm volatile("s_wait_dscnt 0" ::: "memory");

    // ---- online softmax per query row (lanes 0..15, row = lane) ----
    if (lane < 16) {
      const float* Srow = &sS[wv][lane][0];
      float mt = -INFINITY;
#pragma unroll 8
      for (int j = 0; j < 64; ++j) mt = fmaxf(mt, Srow[j]);
      float m_new = fmaxf(m_i, mt);
      float alpha = __expf(m_i - m_new);
      float ts = 0.0f;
#pragma unroll 8
      for (int j = 0; j < 64; ++j) {
        float p = __expf(Srow[j] - m_new);
        ts += p;
        sP[wv][lane][j] = (__bf16)p;
      }
      l_i = l_i * alpha + ts;
      m_i = m_new;
      sScale[wv][lane] = alpha;
    }
    asm volatile("s_wait_dscnt 0" ::: "memory");

    // ---- rescale accumulators by alpha (broadcast via LDS) ----
    float al[8];
#pragma unroll
    for (int r = 0; r < 8; ++r) al[r] = sScale[wv][8 * half + r];
#pragma unroll
    for (int nt = 0; nt < 4; ++nt)
#pragma unroll
      for (int r = 0; r < 8; ++r) o[nt][r] *= al[r];

    // ---- O += P * V ----
    const __bf16* prow = &sP[wv][ln][0];
    bfx16 ap0 = load_fragA(prow, 0, half);
    bfx16 ap1 = load_fragA(prow, 32, half);
#pragma unroll
    for (int nt = 0; nt < 4; ++nt) {
      const __bf16* vcol = Vth + (size_t)(16 * nt + ln) * SEQ + kk0;
      bfx16 bv0 = load_fragB(vcol, 0, half);
      bfx16 bv1 = load_fragB(vcol, 32, half);
      o[nt] = wmma_bf16(ap0, bv0, o[nt]);
      o[nt] = wmma_bf16(ap1, bv1, o[nt]);
    }
  }

  // ---- normalize by 1/l and store to C[B,S,H*Dh] as bf16 ----
  if (lane < 16) sScale[wv][lane] = 1.0f / l_i;
  asm volatile("s_wait_dscnt 0" ::: "memory");
  float inv[8];
#pragma unroll
  for (int r = 0; r < 8; ++r) inv[r] = sScale[wv][8 * half + r];
#pragma unroll
  for (int nt = 0; nt < 4; ++nt) {
#pragma unroll
    for (int r = 0; r < 8; ++r) {
      int gq  = q0 + 8 * half + r;
      int col = h * HDIM + 16 * nt + ln;
      C[((size_t)b * SEQ + gq) * DMODEL + col] = (__bf16)(o[nt][r] * inv[r]);
    }
  }
}

// ---------------------------------------------------------------------------
// Launch: convert -> QKV GEMMs -> attention -> output projection (+bias)
// ---------------------------------------------------------------------------
extern "C" void kernel_launch(void* const* d_in, const int* in_sizes, int n_in,
                              void* d_out, int out_size, void* d_ws, size_t ws_size,
                              hipStream_t stream) {
  (void)in_sizes; (void)n_in; (void)out_size; (void)ws_size;
  const float* X     = (const float*)d_in[0];
  const float* W_K   = (const float*)d_in[1];
  const float* W_Q   = (const float*)d_in[2];
  const float* W_V   = (const float*)d_in[3];
  const float* W_out = (const float*)d_in[4];
  const float* b_out = (const float*)d_in[5];

  char* ws = (char*)d_ws;
  size_t off = 0;
  auto carve = [&](size_t bytes) -> char* {
    char* p = ws + off;
    off += (bytes + 255) & ~(size_t)255;
    return p;
  };
  __bf16* Xbf = (__bf16*)carve((size_t)MTOT * DMODEL * 2);       // 16 MiB
  __bf16* WqT = (__bf16*)carve((size_t)DMODEL * DMODEL * 2);     //  2 MiB
  __bf16* WkT = (__bf16*)carve((size_t)DMODEL * DMODEL * 2);
  __bf16* WvT = (__bf16*)carve((size_t)DMODEL * DMODEL * 2);
  __bf16* WoT = (__bf16*)carve((size_t)DMODEL * DMODEL * 2);
  __bf16* Qb  = (__bf16*)carve((size_t)MTOT * DMODEL * 2);       // [B,H,S,Dh]
  __bf16* Kb  = (__bf16*)carve((size_t)MTOT * DMODEL * 2);       // [B,H,S,Dh]
  __bf16* Vtb = (__bf16*)carve((size_t)MTOT * DMODEL * 2);       // [B,H,Dh,S]
  __bf16* Cb  = (__bf16*)carve((size_t)MTOT * DMODEL * 2);       // [B,S,H*Dh]

  // 1) precision conversion / weight transposition
  cvt_f32_bf16<<<dim3((MTOT * DMODEL) / 1024), 256, 0, stream>>>(X, Xbf, MTOT * DMODEL);
  transpose_w<<<dim3((DMODEL * DMODEL) / 256), 256, 0, stream>>>(W_Q, WqT, DMODEL, DMODEL);
  transpose_w<<<dim3((DMODEL * DMODEL) / 256), 256, 0, stream>>>(W_K, WkT, DMODEL, DMODEL);
  transpose_w<<<dim3((DMODEL * DMODEL) / 256), 256, 0, stream>>>(W_V, WvT, DMODEL, DMODEL);
  transpose_w<<<dim3((DMODEL * DMODEL) / 256), 256, 0, stream>>>(W_out, WoT, DMODEL, DMODEL);

  // 2) QKV projections (WMMA + async-LDS B staging)
  dim3 ggrid(DMODEL / 128, MTOT / 128);
  gemm_bf16<0><<<ggrid, 256, 0, stream>>>(Xbf, WqT, Qb, nullptr, MTOT, DMODEL, DMODEL);
  gemm_bf16<0><<<ggrid, 256, 0, stream>>>(Xbf, WkT, Kb, nullptr, MTOT, DMODEL, DMODEL);
  gemm_bf16<1><<<ggrid, 256, 0, stream>>>(Xbf, WvT, Vtb, nullptr, MTOT, DMODEL, DMODEL);

  // 3) causal attention (WMMA + LDS softmax); 8192 waves, 4 waves/block
  attn_causal<<<dim3(2048), 128, 0, stream>>>(Qb, Kb, Vtb, Cb);

  // 4) output projection + bias (WMMA, f32 out)
  gemm_bf16<2><<<ggrid, 256, 0, stream>>>(Cb, WoT, d_out, b_out, MTOT, DMODEL, DMODEL);
}
